// ContinuousMessagePassing_29703993819530
// MI455X (gfx1250) — matrix operand order repaired and checked by reading, not compiled, not run on hardware
//
#include <hip/hip_runtime.h>
#include <hip/hip_bf16.h>
#include <math.h>

// ---------------------------------------------------------------------------
// Problem constants (all tile-divisible: 50000 = 3125*16, feats mult of 16)
// ---------------------------------------------------------------------------
#define N_NODES   50000
#define N_EDGES   800000
#define IN_FEATS  256
#define OUT_FEATS 256
#define HIDDEN    256
#define MSG       128

// Workspace layout (bytes, all 512-aligned)
#define OFF_W1B   ((size_t)0)                    // 256*256 bf16   = 131072
#define OFF_W2B   ((size_t)131072)               // 128*256 bf16   =  65536
#define OFF_WIHB  ((size_t)196608)               // 768*384 bf16   = 589824
#define OFF_WHHB  ((size_t)786432)               // 768*256 bf16   = 393216
#define OFF_M     ((size_t)1179648)              // 50000*128 f32  = 25600000
#define OFF_SUM   ((size_t)26779648)             // 50000*128 f32  = 25600000
#define OFF_CNT   ((size_t)52379648)             // 50000 f32      = 200000
#define OFF_XB    ((size_t)52579840)             // 50000*256 bf16 = 25600000
#define OFF_ZB    ((size_t)78179840)             // 50000*256 bf16 = 25600000
#define OFF_YB    ((size_t)103779840)            // 50000*128 bf16 = 12800000
                                                 // total ~116.6 MB

typedef __attribute__((ext_vector_type(16))) __bf16        v16bf;
typedef __attribute__((ext_vector_type(8)))  float         v8f;
typedef __attribute__((ext_vector_type(8)))  unsigned int  v8u;
typedef __attribute__((ext_vector_type(4)))  unsigned int  v4u;

// ---------------------------------------------------------------------------
// bf16 helpers (round-to-nearest-even truncation)
// ---------------------------------------------------------------------------
__device__ __forceinline__ unsigned short f2bf(float f) {
  unsigned u = __builtin_bit_cast(unsigned, f);
  return (unsigned short)((u + 0x7FFFu + ((u >> 16) & 1u)) >> 16);
}
__device__ __forceinline__ unsigned pack2bf(float lo, float hi) {
  return (unsigned)f2bf(lo) | ((unsigned)f2bf(hi) << 16);
}

// ---------------------------------------------------------------------------
// WMMA fragment loads (ISA 7.12.2 bf16 layouts, wave32). All sources are
// pre-converted bf16 -> inner loops are pure b128 loads + v_wmma.
// ---------------------------------------------------------------------------
// A-matrix 16x32 bf16 from row-major bf16 in GLOBAL memory.
// lane<16 : row=lane,     K = {k0+0..7, k0+16..23}
// lane>=16: row=lane-16,  K = {k0+8..15, k0+24..31}
__device__ __forceinline__ v16bf load_a_bf16g(const unsigned short* __restrict__ A,
                                              int lda, int m0, int k0) {
  const int lane = threadIdx.x & 31;
  const int row  = m0 + (lane & 15);
  const int hi   = lane >> 4;
  const unsigned short* p = A + (size_t)row * lda + k0 + 8 * hi;
  union { v4u q[2]; v16bf b; } t;
  t.q[0] = *(const v4u*)(p);        // 8 bf16, 16B aligned
  t.q[1] = *(const v4u*)(p + 16);
  return t.b;
}

// A-matrix fragment from bf16 staged in LDS, row-major [16][256]
__device__ __forceinline__ v16bf load_a_lds_bf16(const unsigned short* H, int k0) {
  const int lane = threadIdx.x & 31;
  const int row  = lane & 15;
  const int hi   = lane >> 4;
  const unsigned short* p = H + row * 256 + k0 + 8 * hi;
  union { v4u q[2]; v16bf b; } t;
  t.q[0] = *(const v4u*)(p);
  t.q[1] = *(const v4u*)(p + 16);
  return t.b;
}

// B-matrix 32x16 bf16 from row-major bf16 weight W[N][K] (B = W^T tile).
// lane<16: col=n0+lane,  K = {k0+0..7, k0+16..23} ; lane>=16: col=n0+lane-16, K = {k0+8..15, k0+24..31}
__device__ __forceinline__ v16bf load_b_bf16(const unsigned short* __restrict__ W,
                                             int ldk, int n0, int k0) {
  const int lane = threadIdx.x & 31;
  const int col  = n0 + (lane & 15);
  const int k    = k0 + 8 * (lane >> 4);
  union { v4u q[2]; v16bf b; } t;
  t.q[0] = *(const v4u*)(W + (size_t)col * ldk + k);
  t.q[1] = *(const v4u*)(W + (size_t)col * ldk + k + 16);
  return t.b;
}

__device__ __forceinline__ v8f wmma_bf(v16bf a, v16bf b, v8f c) {
  return __builtin_amdgcn_wmma_f32_16x16x32_bf16(false, a, false, b,
                                                 (short)0, c, false, false);
}

__device__ __forceinline__ float sigmoidf_(float v) {
  return 1.0f / (1.0f + __expf(-v));
}

// ---------------------------------------------------------------------------
// Kernel 1: vectorized f32 -> bf16 conversion (8 elems/thread, b128 in/out)
// count must be a multiple of 8 (all our tensors are)
// ---------------------------------------------------------------------------
__global__ void cvt_bf16x8_kernel(const float* __restrict__ in,
                                  unsigned short* __restrict__ out, int count8) {
  int i = blockIdx.x * blockDim.x + threadIdx.x;
  if (i < count8) {
    const float4* p = (const float4*)(in + (size_t)i * 8);
    float4 a = p[0], b = p[1];
    v4u o;
    o[0] = pack2bf(a.x, a.y); o[1] = pack2bf(a.z, a.w);
    o[2] = pack2bf(b.x, b.y); o[3] = pack2bf(b.z, b.w);
    *(v4u*)(out + (size_t)i * 8) = o;
  }
}

// ---------------------------------------------------------------------------
// Kernel 2: fused encoder  m = relu(relu(x@W1^T + b1)@W2^T + b2)
// one block = 16 nodes, 8 waves; layer-1 output staged bf16 in LDS.
// Layer 1: each wave owns 2 N-tiles and shares each A fragment across both.
// ---------------------------------------------------------------------------
__global__ void __launch_bounds__(256)
encoder_kernel(const unsigned short* __restrict__ xb,
               const unsigned short* __restrict__ W1b, const float* __restrict__ b1,
               const unsigned short* __restrict__ W2b, const float* __restrict__ b2,
               float* __restrict__ m_out) {
  __shared__ unsigned short h_lds[16 * 256];   // 8 KB
  const int m0   = blockIdx.x * 16;
  const int wave = threadIdx.x >> 5;
  const int lane = threadIdx.x & 31;
  const int n_l  = lane & 15;
  const int mb   = 8 * (lane >> 4);

  // layer 1: tiles (wave) and (wave+8); A fragment loaded once per K-step
  {
    const int na = wave * 16;
    const int nb = (wave + 8) * 16;
    v8f acc0 = {}, acc1 = {};
    for (int k0 = 0; k0 < IN_FEATS; k0 += 32) {
      v16bf a = load_a_bf16g(xb, IN_FEATS, m0, k0);
      acc0 = wmma_bf(a, load_b_bf16(W1b, IN_FEATS, na, k0), acc0);
      acc1 = wmma_bf(a, load_b_bf16(W1b, IN_FEATS, nb, k0), acc1);
    }
    const float biasa = b1[na + n_l];
    const float biasb = b1[nb + n_l];
    #pragma unroll
    for (int r = 0; r < 8; ++r) {
      h_lds[(mb + r) * 256 + na + n_l] = f2bf(fmaxf(acc0[r] + biasa, 0.0f));
      h_lds[(mb + r) * 256 + nb + n_l] = f2bf(fmaxf(acc1[r] + biasb, 0.0f));
    }
  }
  __syncthreads();

  // layer 2: 8 output tiles of 16, one per wave; A from LDS
  {
    const int n0 = wave * 16;
    v8f acc = {};
    for (int k0 = 0; k0 < HIDDEN; k0 += 32) {
      v16bf a = load_a_lds_bf16(h_lds, k0);
      v16bf b = load_b_bf16(W2b, HIDDEN, n0, k0);
      acc = wmma_bf(a, b, acc);
    }
    const float bias = b2[n0 + n_l];
    #pragma unroll
    for (int r = 0; r < 8; ++r) {
      float v = fmaxf(acc[r] + bias, 0.0f);
      m_out[(size_t)(m0 + mb + r) * MSG + n0 + n_l] = v;
    }
  }
}

// ---------------------------------------------------------------------------
// Kernel 3: zero the mailbox accumulators (needed every call: graph replay)
// ---------------------------------------------------------------------------
__global__ void zero_kernel(float* __restrict__ p, int count) {
  int i = blockIdx.x * blockDim.x + threadIdx.x;
  if (i < count) p[i] = 0.0f;
}

// ---------------------------------------------------------------------------
// Kernel 4: edge scatter-add. one wave per edge, 4 floats per lane.
// summed[] is 25.6 MB -> L2 resident, atomics resolve at L2.
// ---------------------------------------------------------------------------
__global__ void __launch_bounds__(256)
scatter_kernel(const float* __restrict__ m,
               const int* __restrict__ src, const int* __restrict__ dst,
               float* __restrict__ summed, float* __restrict__ cnt) {
  const int gid  = blockIdx.x * blockDim.x + threadIdx.x;
  const int e    = gid >> 5;
  const int lane = gid & 31;
  if (e >= N_EDGES) return;
  const int s = src[e];
  const int d = dst[e];
  const int f = lane * 4;
  float4 v = *(const float4*)(m + (size_t)s * MSG + f);
  float* o = summed + (size_t)d * MSG + f;
  atomicAdd(o + 0, v.x);
  atomicAdd(o + 1, v.y);
  atomicAdd(o + 2, v.z);
  atomicAdd(o + 3, v.w);
  if (lane == 0) atomicAdd(cnt + d, 1.0f);
}

// ---------------------------------------------------------------------------
// Kernel 5: y_bf = bf16( summed / max(cnt,1) )   (8 elems/thread)
// ---------------------------------------------------------------------------
__global__ void mean_bf16_kernel(const float* __restrict__ summed,
                                 const float* __restrict__ cnt,
                                 unsigned short* __restrict__ y_bf) {
  int i = blockIdx.x * blockDim.x + threadIdx.x;           // group of 8 elems
  if (i < N_NODES * MSG / 8) {
    const int n = (i * 8) >> 7;                            // node index
    const float inv = 1.0f / fmaxf(cnt[n], 1.0f);
    const float4* p = (const float4*)(summed + (size_t)i * 8);
    float4 a = p[0], b = p[1];
    v4u o;
    o[0] = pack2bf(a.x * inv, a.y * inv); o[1] = pack2bf(a.z * inv, a.w * inv);
    o[2] = pack2bf(b.x * inv, b.y * inv); o[3] = pack2bf(b.z * inv, b.w * inv);
    *(v4u*)(y_bf + (size_t)i * 8) = o;
  }
}

// ---------------------------------------------------------------------------
// Kernel 6: fused GRU cell. Each wave owns a 16-node x 16-feature output tile
// and 6 accumulators (r/u/n gates for gi and gh); each A fragment feeds 3
// WMMAs. Inner loops: b128 loads + v_wmma only. grid = (3125, 2), block 256.
// ---------------------------------------------------------------------------
__global__ void __launch_bounds__(256)
gru_kernel(const unsigned short* __restrict__ xb,
           const unsigned short* __restrict__ yb,
           const unsigned short* __restrict__ zb,
           const float* __restrict__ z,
           const unsigned short* __restrict__ Wihb, const float* __restrict__ b_ih,
           const unsigned short* __restrict__ Whhb, const float* __restrict__ b_hh,
           float* __restrict__ out) {
  const int m0   = blockIdx.x * 16;
  const int wave = threadIdx.x >> 5;
  const int lane = threadIdx.x & 31;
  const int n0   = (blockIdx.y * 8 + wave) * 16;  // final-feature tile

  v8f gir = {}, giu = {}, gin = {};
  v8f ghr = {}, ghu = {}, ghn = {};

  // gi: x part (K = 0..255 of W_ih, ldk = 384)
  for (int k0 = 0; k0 < IN_FEATS; k0 += 32) {
    v16bf a = load_a_bf16g(xb, IN_FEATS, m0, k0);
    gir = wmma_bf(a, load_b_bf16(Wihb, 384, n0, k0), gir);
    giu = wmma_bf(a, load_b_bf16(Wihb, 384, OUT_FEATS + n0, k0), giu);
    gin = wmma_bf(a, load_b_bf16(Wihb, 384, 2 * OUT_FEATS + n0, k0), gin);
  }
  // gi: y part (K = 256..383 of W_ih)
  for (int k0 = 0; k0 < MSG; k0 += 32) {
    v16bf a = load_a_bf16g(yb, MSG, m0, k0);
    gir = wmma_bf(a, load_b_bf16(Wihb, 384, n0, IN_FEATS + k0), gir);
    giu = wmma_bf(a, load_b_bf16(Wihb, 384, OUT_FEATS + n0, IN_FEATS + k0), giu);
    gin = wmma_bf(a, load_b_bf16(Wihb, 384, 2 * OUT_FEATS + n0, IN_FEATS + k0), gin);
  }
  // gh: z @ W_hh^T (ldk = 256)
  for (int k0 = 0; k0 < OUT_FEATS; k0 += 32) {
    v16bf a = load_a_bf16g(zb, OUT_FEATS, m0, k0);
    ghr = wmma_bf(a, load_b_bf16(Whhb, OUT_FEATS, n0, k0), ghr);
    ghu = wmma_bf(a, load_b_bf16(Whhb, OUT_FEATS, OUT_FEATS + n0, k0), ghu);
    ghn = wmma_bf(a, load_b_bf16(Whhb, OUT_FEATS, 2 * OUT_FEATS + n0, k0), ghn);
  }

  const int n  = n0 + (lane & 15);
  const int mb = 8 * (lane >> 4);
  const float bir = b_ih[n], biu = b_ih[OUT_FEATS + n], bin = b_ih[2 * OUT_FEATS + n];
  const float bhr = b_hh[n], bhu = b_hh[OUT_FEATS + n], bhn = b_hh[2 * OUT_FEATS + n];

  #pragma unroll
  for (int r = 0; r < 8; ++r) {
    const int node = m0 + mb + r;
    const float zz = z[(size_t)node * OUT_FEATS + n];   // fp32 z for the blend
    const float rg = sigmoidf_((gir[r] + bir) + (ghr[r] + bhr));
    const float ug = sigmoidf_((giu[r] + biu) + (ghu[r] + bhu));
    const float ng = tanhf((gin[r] + bin) + rg * (ghn[r] + bhn));
    const float ho = (1.0f - ug) * ng + ug * zz;
    out[(size_t)node * OUT_FEATS + n] = ho;
    out[(size_t)N_NODES * OUT_FEATS + (size_t)node * OUT_FEATS + n] = ho;  // tuple copy
  }
}

// ---------------------------------------------------------------------------
// Host launcher
// ---------------------------------------------------------------------------
extern "C" void kernel_launch(void* const* d_in, const int* in_sizes, int n_in,
                              void* d_out, int out_size, void* d_ws, size_t ws_size,
                              hipStream_t stream) {
  const float* x    = (const float*)d_in[0];
  const float* z    = (const float*)d_in[1];
  const int*   src  = (const int*)d_in[2];
  const int*   dst  = (const int*)d_in[3];
  const float* W1   = (const float*)d_in[4];
  const float* b1   = (const float*)d_in[5];
  const float* W2   = (const float*)d_in[6];
  const float* b2   = (const float*)d_in[7];
  const float* Wih  = (const float*)d_in[8];
  const float* bih  = (const float*)d_in[9];
  const float* Whh  = (const float*)d_in[10];
  const float* bhh  = (const float*)d_in[11];

  char* ws = (char*)d_ws;
  unsigned short* W1b  = (unsigned short*)(ws + OFF_W1B);
  unsigned short* W2b  = (unsigned short*)(ws + OFF_W2B);
  unsigned short* Wihb = (unsigned short*)(ws + OFF_WIHB);
  unsigned short* Whhb = (unsigned short*)(ws + OFF_WHHB);
  float* m      = (float*)(ws + OFF_M);
  float* summed = (float*)(ws + OFF_SUM);
  float* cnt    = (float*)(ws + OFF_CNT);
  unsigned short* xb = (unsigned short*)(ws + OFF_XB);
  unsigned short* zb = (unsigned short*)(ws + OFF_ZB);
  unsigned short* yb = (unsigned short*)(ws + OFF_YB);
  float* out    = (float*)d_out;

  // bf16 conversions (vectorized; one-shot, then everything is L2/HBM bf16)
  #define CVT8(IN, OUT, COUNT) \
    cvt_bf16x8_kernel<<<((COUNT) / 8 + 255) / 256, 256, 0, stream>>>(IN, OUT, (COUNT) / 8)
  CVT8(W1,  W1b,  HIDDEN * IN_FEATS);
  CVT8(W2,  W2b,  MSG * HIDDEN);
  CVT8(Wih, Wihb, 3 * OUT_FEATS * (IN_FEATS + MSG));
  CVT8(Whh, Whhb, 3 * OUT_FEATS * OUT_FEATS);
  CVT8(x,   xb,   N_NODES * IN_FEATS);
  CVT8(z,   zb,   N_NODES * OUT_FEATS);
  #undef CVT8

  // zero mailbox (summed + cnt are contiguous in ws)
  const int zcount = N_NODES * MSG + N_NODES;
  zero_kernel<<<(zcount + 255) / 256, 256, 0, stream>>>(summed, zcount);

  // fused 2-layer encoder -> m
  encoder_kernel<<<N_NODES / 16, 256, 0, stream>>>(xb, W1b, b1, W2b, b2, m);

  // edge scatter-add (wave per edge)
  scatter_kernel<<<(N_EDGES * 32) / 256, 256, 0, stream>>>(m, src, dst, summed, cnt);

  // mailbox mean -> bf16 y
  mean_bf16_kernel<<<(N_NODES * MSG / 8 + 255) / 256, 256, 0, stream>>>(summed, cnt, yb);

  // fused GRU
  gru_kernel<<<dim3(N_NODES / 16, 2), 256, 0, stream>>>(xb, yb, zb, z,
                                                        Wihb, bih, Whhb, bhh, out);
}
